// MistralAttention_68582037783154
// MI455X (gfx1250) — matrix operand order repaired and checked by reading, not compiled, not run on hardware
//
#include <hip/hip_runtime.h>
#include <hip/hip_bf16.h>

// ---------------------------------------------------------------------------
// Mistral GQA attention block for gfx1250 (MI455X), bf16 WMMA pipeline:
//   1) convert X + transpose-convert weights to bf16 (1/sqrt(d) folded
//      into Wq so attention scores come out of WMMA pre-scaled)
//   2) Q/K/V projections via wmma_f32_16x16x32_bf16 GEMM
//      (V written pre-transposed per batch for async attention staging)
//   3) RoPE applied in-place on Q,K (f32 math, bf16 storage)
//   4) flash attention (online softmax, causal), double-buffered
//      global_load_async_to_lds_b128 K/V staging, row-sums via WMMA,
//      mask-free fast path for tiles below the causal diagonal
//   5) output projection -> f32 d_out
// ---------------------------------------------------------------------------

typedef __bf16 bf16_t;
typedef __attribute__((ext_vector_type(16))) __bf16 v16bf;
typedef __attribute__((ext_vector_type(8)))  __bf16 v8bf;
typedef __attribute__((ext_vector_type(8)))  float  v8f;

#define HIDDEN 4096
#define NH     32
#define NKV    8
#define HD     128
#define SEQ    2048
#define BSZ    2
#define KVD    (NKV * HD)   // 1024

struct FalseT { static constexpr bool value = false; };
struct TrueT  { static constexpr bool value = true;  };

__device__ __forceinline__ v8f zero8() {
  v8f z = {0.f, 0.f, 0.f, 0.f, 0.f, 0.f, 0.f, 0.f};
  return z;
}

__device__ __forceinline__ v8f wmma_bf16(v16bf a, v16bf b, v8f c) {
  return __builtin_amdgcn_wmma_f32_16x16x32_bf16(
      /*neg_a=*/false, a, /*neg_b=*/false, b,
      /*c_mod=*/(short)0, c, /*reuse_a=*/false, /*reuse_b=*/false);
}

// Async global -> LDS copy, 16 bytes per lane. GV addressing (vaddr64, off).
// Tracked by ASYNCcnt; LDS dest byte offset = low 32 bits of generic pointer.
__device__ __forceinline__ void async_b128(unsigned lds_off, const void* g) {
  asm volatile("global_load_async_to_lds_b128 %0, %1, off"
               :: "v"(lds_off), "v"((unsigned long long)(size_t)g)
               : "memory");
}
__device__ __forceinline__ void wait_async0() {
  asm volatile("s_wait_asynccnt 0x0" ::: "memory");
}

// A fragment (16x32 bf16, row-major source, arbitrary lda).
// Layout: lanes 0-15 row M=lane, K 0..15;
//         lanes 16-31 same rows, K 16..31 (matches B fragments).
__device__ __forceinline__ v16bf load_a_frag(const bf16_t* a, int lda) {
  const int l = threadIdx.x & 31;
  const bf16_t* p = a + (size_t)(l & 15) * lda + ((l >> 4) << 4);
  v8bf lo = *(const v8bf*)p;
  v8bf hi = *(const v8bf*)(p + 8);
  v16bf r;
#pragma unroll
  for (int i = 0; i < 8; ++i) { r[i] = lo[i]; r[i + 8] = hi[i]; }
  return r;
}

// B fragment (32x16 bf16) sourced from B^T stored row-major (N x K, ldb):
// lane holds column N = lane&15; halves are 16 contiguous K values at
// K-offset (lane>>4)*16  -> two 16B loads.
__device__ __forceinline__ v16bf load_bt_frag(const bf16_t* bt, int ldb) {
  const int l = threadIdx.x & 31;
  const bf16_t* p = bt + (size_t)(l & 15) * ldb + ((l >> 4) << 4);
  v8bf lo = *(const v8bf*)p;
  v8bf hi = *(const v8bf*)(p + 8);
  v16bf r;
#pragma unroll
  for (int i = 0; i < 8; ++i) { r[i] = lo[i]; r[i + 8] = hi[i]; }
  return r;
}

// ---------------------------------------------------------------------------
// f32 -> bf16 conversion (grid-stride)
// ---------------------------------------------------------------------------
__global__ __launch_bounds__(256) void cvt_kernel(const float* __restrict__ src,
                                                  bf16_t* __restrict__ dst,
                                                  size_t n) {
  size_t i = (size_t)blockIdx.x * blockDim.x + threadIdx.x;
  const size_t stride = (size_t)gridDim.x * blockDim.x;
  for (; i < n; i += stride) dst[i] = (bf16_t)src[i];
}

// W (K x N f32, row-major) -> WT (N x K bf16, row-major) * scale.
__global__ __launch_bounds__(256) void transpose_cvt_kernel(
    const float* __restrict__ W, bf16_t* __restrict__ WT, int K, int N,
    float scale) {
  __shared__ float tile[32][33];
  const int kb = blockIdx.y * 32, nb = blockIdx.x * 32;
  const int tx = threadIdx.x & 31, ty = threadIdx.x >> 5;
#pragma unroll
  for (int i = ty; i < 32; i += 8)
    tile[i][tx] = W[(size_t)(kb + i) * N + (nb + tx)];
  __syncthreads();
#pragma unroll
  for (int i = ty; i < 32; i += 8)
    WT[(size_t)(nb + i) * K + (kb + tx)] = (bf16_t)(tile[tx][i] * scale);
}

// ---------------------------------------------------------------------------
// GEMM: C(MxN) = A(MxK, bf16 row-major) * B(KxN), B supplied as BT (NxK).
// Block = 256 threads = 8 waves, block tile 64x128, wave tile 32x32
// (2x2 WMMA frags), K-step 32, f32 accumulate, global_prefetch lookahead.
// MODE: 0 = f32 row-major out, 1 = bf16 row-major out,
//       2 = bf16 transposed-per-batch out  C[b][col][s]  (for V)
// ---------------------------------------------------------------------------
template <int MODE>
__global__ __launch_bounds__(256) void gemm_bf16_tn(
    const bf16_t* __restrict__ A, const bf16_t* __restrict__ BT,
    void* __restrict__ Cout, int M, int N, int K) {
  const int wave = threadIdx.x >> 5;
  const int m0 = blockIdx.y * 64 + (wave & 1) * 32;
  const int n0 = blockIdx.x * 128 + (wave >> 1) * 32;
  const int l = threadIdx.x & 31;

  v8f c00 = zero8(), c01 = zero8(), c10 = zero8(), c11 = zero8();

  const bf16_t* a0p = A + (size_t)m0 * K;
  const bf16_t* a1p = A + (size_t)(m0 + 16) * K;
  const bf16_t* b0p = BT + (size_t)n0 * K;
  const bf16_t* b1p = BT + (size_t)(n0 + 16) * K;

  for (int k = 0; k < K; k += 32) {
    __builtin_prefetch(a0p + k + 512, 0, 3);  // global_prefetch_b8, 16 steps
    __builtin_prefetch(b0p + k + 512, 0, 3);
    v16bf a0 = load_a_frag(a0p + k, K);
    v16bf a1 = load_a_frag(a1p + k, K);
    v16bf b0 = load_bt_frag(b0p + k, K);
    v16bf b1 = load_bt_frag(b1p + k, K);
    c00 = wmma_bf16(a0, b0, c00);
    c01 = wmma_bf16(a0, b1, c01);
    c10 = wmma_bf16(a1, b0, c10);
    c11 = wmma_bf16(a1, b1, c11);
  }

  // D layout: lane&15 = column, (lane>>4)*8 + r = row.
  const int col = l & 15;
  const int rb = (l >> 4) * 8;
  if (MODE == 0) {
    float* C = (float*)Cout;
#pragma unroll
    for (int r = 0; r < 8; ++r) {
      C[(size_t)(m0 + rb + r) * N + n0 + col]           = c00[r];
      C[(size_t)(m0 + rb + r) * N + n0 + 16 + col]      = c01[r];
      C[(size_t)(m0 + 16 + rb + r) * N + n0 + col]      = c10[r];
      C[(size_t)(m0 + 16 + rb + r) * N + n0 + 16 + col] = c11[r];
    }
  } else if (MODE == 1) {
    bf16_t* C = (bf16_t*)Cout;
#pragma unroll
    for (int r = 0; r < 8; ++r) {
      C[(size_t)(m0 + rb + r) * N + n0 + col]           = (bf16_t)c00[r];
      C[(size_t)(m0 + rb + r) * N + n0 + 16 + col]      = (bf16_t)c01[r];
      C[(size_t)(m0 + 16 + rb + r) * N + n0 + col]      = (bf16_t)c10[r];
      C[(size_t)(m0 + 16 + rb + r) * N + n0 + 16 + col] = (bf16_t)c11[r];
    }
  } else {
    // C[b][col][s], row m = b*SEQ + s, N == KVD
    bf16_t* C = (bf16_t*)Cout;
#pragma unroll
    for (int r = 0; r < 8; ++r) {
      const int mr0 = m0 + rb + r, mr1 = m0 + 16 + rb + r;
      const int b0i = mr0 >> 11, s0i = mr0 & (SEQ - 1);
      const int b1i = mr1 >> 11, s1i = mr1 & (SEQ - 1);
      C[((size_t)b0i * KVD + n0 + col) * SEQ + s0i]      = (bf16_t)c00[r];
      C[((size_t)b0i * KVD + n0 + 16 + col) * SEQ + s0i] = (bf16_t)c01[r];
      C[((size_t)b1i * KVD + n0 + col) * SEQ + s1i]      = (bf16_t)c10[r];
      C[((size_t)b1i * KVD + n0 + 16 + col) * SEQ + s1i] = (bf16_t)c11[r];
    }
  }
}

// ---------------------------------------------------------------------------
// RoPE in-place on Q [B,S,NH,HD] and K [B,S,NKV,HD] (bf16).
// ---------------------------------------------------------------------------
__global__ __launch_bounds__(256) void rope_kernel(bf16_t* __restrict__ Qb,
                                                   bf16_t* __restrict__ Kb) {
  size_t idx = (size_t)blockIdx.x * blockDim.x + threadIdx.x;
  const int j = (int)(idx & 63); idx >>= 6;
  const int hh = (int)(idx % (NH + NKV)); idx /= (NH + NKV);
  const int s = (int)(idx % SEQ);
  const int b = (int)(idx / SEQ);
  if (b >= BSZ) return;

  // inv_freq[j] = theta^(-2j/128) = exp(-j * ln(10000)/64)
  const float invf = __expf(-(float)j * 0.14391156831212787f);
  const float ang = (float)s * invf;
  float sn, cs;
  __sincosf(ang, &sn, &cs);

  bf16_t* base = (hh < NH)
      ? (Qb + ((size_t)b * SEQ + s) * HIDDEN + (size_t)hh * HD)
      : (Kb + ((size_t)b * SEQ + s) * KVD + (size_t)(hh - NH) * HD);
  const float x1 = (float)base[j];
  const float x2 = (float)base[j + 64];
  base[j]      = (bf16_t)(x1 * cs - x2 * sn);
  base[j + 64] = (bf16_t)(x2 * cs + x1 * sn);
}

// ---------------------------------------------------------------------------
// Flash attention, causal. Grid (SEQ/128, NH, BSZ), block 256 = 8 waves.
// One (b,head) per block, 128 query rows, 16 per wave. K/V 32-key tiles
// double-buffered in LDS via global_load_async_to_lds_b128 (ASYNCcnt).
// V arrives pre-transposed from global (Vt[b][d][s]). Row sums via WMMA
// against an all-ones B fragment; row max via half-wave shfl butterfly.
// Scores are pre-scaled (1/sqrt(d) folded into Wq); tiles strictly below
// the causal diagonal take a compile-time mask-free fast path.
// ---------------------------------------------------------------------------
#define LDS_K_STRIDE 136   // 32 keys x 128 d, padded rows (272B)
#define LDS_V_STRIDE 40    // 128 d x 32 keys, padded rows (80B)
#define LDS_P_STRIDE 40    // per-wave 16 x 32 P tile, padded

__global__ __launch_bounds__(256) void flash_attn_kernel(
    const bf16_t* __restrict__ Q, const bf16_t* __restrict__ Kmat,
    const bf16_t* __restrict__ Vt, bf16_t* __restrict__ O) {
  __shared__ bf16_t sK[2][32 * LDS_K_STRIDE];
  __shared__ bf16_t sVt[2][128 * LDS_V_STRIDE];
  __shared__ bf16_t sP[8 * 16 * LDS_P_STRIDE];

  const int head = blockIdx.y;
  const int b = blockIdx.z;
  const int kvh = head >> 2;                 // N_GROUPS = 4
  const int tid = threadIdx.x;
  const int wave = tid >> 5;
  const int lane = tid & 31;
  const int lcol = lane & 15;
  const int lhalf = lane >> 4;
  const int q0 = blockIdx.x * 128;
  const int qrow = q0 + wave * 16;

  // Per-thread async-copy geometry (2 x 16B chunks each for K and V).
  const int kc0 = tid, kc1 = tid + 256;      // K tile: 512 chunks (32x16)
  const bf16_t* kbase = Kmat + ((size_t)b * SEQ) * KVD + (size_t)kvh * HD;
  const bf16_t* vbase = Vt + ((size_t)b * KVD + (size_t)kvh * HD) * SEQ;

  // Q fragments: 16 rows x 128 d, four 16x32 A-frags (lda = HIDDEN).
  const bf16_t* qp = Q + ((size_t)b * SEQ + qrow) * HIDDEN + (size_t)head * HD;
  v16bf qf[4];
#pragma unroll
  for (int c = 0; c < 4; ++c) qf[c] = load_a_frag(qp + c * 32, HIDDEN);

  // All-ones B fragment for WMMA row sums.
  v16bf ones;
#pragma unroll
  for (int i = 0; i < 16; ++i) ones[i] = (bf16_t)1.0f;

  v8f o[8];
  float mrow[8], lrow[8];
#pragma unroll
  for (int f = 0; f < 8; ++f) o[f] = zero8();
#pragma unroll
  for (int r = 0; r < 8; ++r) { mrow[r] = -1e30f; lrow[r] = 0.f; }

  const int ntiles = q0 / 32 + 4;  // causal bound, 32 keys/tile
  const int nfull = q0 / 32;       // tiles fully below diagonal: no mask

  auto issue_tile = [&](int buf, int kbeg) {
#pragma unroll
    for (int i = 0; i < 2; ++i) {
      const int ck = (i == 0) ? kc0 : kc1;
      const int krow = ck >> 4, kseg = (ck & 15) * 8;
      async_b128((unsigned)(size_t)&sK[buf][krow * LDS_K_STRIDE + kseg],
                 kbase + (size_t)(kbeg + krow) * KVD + kseg);
      const int vrow = ck >> 2, vseg = (ck & 3) * 8;
      async_b128((unsigned)(size_t)&sVt[buf][vrow * LDS_V_STRIDE + vseg],
                 vbase + (size_t)vrow * SEQ + kbeg + vseg);
    }
  };

  auto tile_body = [&](int kt, auto masked_t) {
    constexpr bool MASKED = decltype(masked_t)::value;
    const int cur = kt & 1;
    const int kbeg = kt * 32;
    wait_async0();        // this wave's async copies for tile kt complete
    __syncthreads();      // => every wave's copies complete; prev buf free
    if (kt + 1 < ntiles) issue_tile(1 - cur, kbeg + 32);

    // ---- scores: S(16x32) = Q(16x128) * K_tile^T, two 16x16 frags
    v8f s0 = zero8(), s1 = zero8();
#pragma unroll
    for (int c = 0; c < 4; ++c) {
      const bf16_t* p0 = &sK[cur][lcol * LDS_K_STRIDE + c * 32 + lhalf * 16];
      v8bf b0lo = *(const v8bf*)p0;
      v8bf b0hi = *(const v8bf*)(p0 + 8);
      const bf16_t* p1 =
          &sK[cur][(16 + lcol) * LDS_K_STRIDE + c * 32 + lhalf * 16];
      v8bf b1lo = *(const v8bf*)p1;
      v8bf b1hi = *(const v8bf*)(p1 + 8);
      v16bf bf0, bf1;
#pragma unroll
      for (int i = 0; i < 8; ++i) {
        bf0[i] = b0lo[i]; bf0[i + 8] = b0hi[i];
        bf1[i] = b1lo[i]; bf1[i + 8] = b1hi[i];
      }
      s0 = wmma_bf16(qf[c], bf0, s0);
      s1 = wmma_bf16(qf[c], bf1, s1);
    }

    // ---- online softmax (scores pre-scaled via Wq)
    const int k0abs = kbeg + lcol;
    const int k1abs = k0abs + 16;
    float alpha[8];
    float p0v[8], p1v[8];
#pragma unroll
    for (int r = 0; r < 8; ++r) {
      float x0, x1;
      if (MASKED) {
        const int ra = qrow + lhalf * 8 + r;
        x0 = s0[r] + ((k0abs <= ra) ? 0.f : -1e9f);
        x1 = s1[r] + ((k1abs <= ra) ? 0.f : -1e9f);
      } else {
        x0 = s0[r];
        x1 = s1[r];
      }
      float mt = fmaxf(x0, x1);
#pragma unroll
      for (int off = 1; off < 16; off <<= 1)
        mt = fmaxf(mt, __shfl_xor(mt, off, 16));
      const float mnew = fmaxf(mrow[r], mt);
      alpha[r] = __expf(mrow[r] - mnew);
      mrow[r] = mnew;
      p0v[r] = __expf(x0 - mnew);
      p1v[r] = __expf(x1 - mnew);
    }

    // ---- P (f32 D-layout) -> LDS -> bf16 A-layout (wave-private tile;
    //      DS ops are in-order per wave on CDNA5, fence for scheduling)
    {
      const int rowb = wave * 16 + lhalf * 8;
#pragma unroll
      for (int r = 0; r < 8; ++r) {
        sP[(rowb + r) * LDS_P_STRIDE + lcol] = (bf16_t)p0v[r];
        sP[(rowb + r) * LDS_P_STRIDE + 16 + lcol] = (bf16_t)p1v[r];
      }
    }
    asm volatile("s_wait_dscnt 0x0" ::: "memory");
    v16bf pf;
    {
      const bf16_t* pp = &sP[(wave * 16 + lcol) * LDS_P_STRIDE + lhalf * 16];
      v8bf lo = *(const v8bf*)pp;
      v8bf hi = *(const v8bf*)(pp + 8);
#pragma unroll
      for (int i = 0; i < 8; ++i) { pf[i] = lo[i]; pf[i + 8] = hi[i]; }
    }

    // ---- row sums via WMMA: srow[r] = sum_k P[row=r, k] (all lanes)
    v8f srow = wmma_bf16(pf, ones, zero8());
#pragma unroll
    for (int r = 0; r < 8; ++r) {
      lrow[r] = lrow[r] * alpha[r] + srow[r];
#pragma unroll
      for (int f = 0; f < 8; ++f) o[f][r] *= alpha[r];
    }

    // ---- O += P(16x32) * V_tile(32x128), 8 output d-frags
#pragma unroll
    for (int f = 0; f < 8; ++f) {
      const bf16_t* vp2 =
          &sVt[cur][(f * 16 + lcol) * LDS_V_STRIDE + lhalf * 16];
      v8bf lo = *(const v8bf*)vp2;
      v8bf hi = *(const v8bf*)(vp2 + 8);
      v16bf vf;
#pragma unroll
      for (int i = 0; i < 8; ++i) { vf[i] = lo[i]; vf[i + 8] = hi[i]; }
      o[f] = wmma_bf16(pf, vf, o[f]);
    }
  };

  issue_tile(0, 0);
  for (int kt = 0; kt < nfull; ++kt) tile_body(kt, FalseT{});  // no mask
  for (int kt = nfull; kt < ntiles; ++kt) tile_body(kt, TrueT{});  // diagonal

  // ---- epilogue: normalize and store O as bf16 in [B,S,NH,HD]
#pragma unroll
  for (int r = 0; r < 8; ++r) {
    const float inv = 1.f / lrow[r];
    const int row = qrow + lhalf * 8 + r;
    bf16_t* op = O + ((size_t)b * SEQ + row) * HIDDEN + (size_t)head * HD;
#pragma unroll
    for (int f = 0; f < 8; ++f)
      op[f * 16 + lcol] = (bf16_t)(o[f][r] * inv);
  }
}

// ---------------------------------------------------------------------------
// Host-side orchestration
// ---------------------------------------------------------------------------
extern "C" void kernel_launch(void* const* d_in, const int* in_sizes, int n_in,
                              void* d_out, int out_size, void* d_ws,
                              size_t ws_size, hipStream_t stream) {
  (void)in_sizes; (void)n_in; (void)out_size; (void)ws_size;
  const float* X  = (const float*)d_in[0];
  // d_in[1] attention_mask (causal, reproduced in-kernel)
  // d_in[2] position_ids   (arange, reproduced in-kernel)
  const float* Wq = (const float*)d_in[3];
  const float* Wk = (const float*)d_in[4];
  const float* Wv = (const float*)d_in[5];
  const float* Wo = (const float*)d_in[6];

  char* ws = (char*)d_ws;
  size_t off = 0;
  auto alloc = [&](size_t bytes) -> void* {
    void* p = ws + off;
    off = (off + bytes + 255) & ~(size_t)255;
    return p;
  };

  const size_t MS = (size_t)BSZ * SEQ;  // 4096 rows
  bf16_t* Xb  = (bf16_t*)alloc(MS * HIDDEN * sizeof(bf16_t));
  bf16_t* WqT = (bf16_t*)alloc((size_t)HIDDEN * HIDDEN * sizeof(bf16_t));
  bf16_t* WkT = (bf16_t*)alloc((size_t)KVD * HIDDEN * sizeof(bf16_t));
  bf16_t* WvT = (bf16_t*)alloc((size_t)KVD * HIDDEN * sizeof(bf16_t));
  bf16_t* WoT = (bf16_t*)alloc((size_t)HIDDEN * HIDDEN * sizeof(bf16_t));
  bf16_t* Qb  = (bf16_t*)alloc(MS * HIDDEN * sizeof(bf16_t));
  bf16_t* Kb  = (bf16_t*)alloc(MS * KVD * sizeof(bf16_t));
  bf16_t* Vtr = (bf16_t*)alloc(MS * KVD * sizeof(bf16_t));  // [b][col][s]
  bf16_t* Ob  = (bf16_t*)alloc(MS * HIDDEN * sizeof(bf16_t));

  // 1) bf16 staging; fold 1/sqrt(HD) into Wq
  cvt_kernel<<<4096, 256, 0, stream>>>(X, Xb, MS * HIDDEN);
  transpose_cvt_kernel<<<dim3(HIDDEN / 32, HIDDEN / 32), 256, 0, stream>>>(
      Wq, WqT, HIDDEN, HIDDEN, 0.08838834764831845f);
  transpose_cvt_kernel<<<dim3(KVD / 32, HIDDEN / 32), 256, 0, stream>>>(
      Wk, WkT, HIDDEN, KVD, 1.0f);
  transpose_cvt_kernel<<<dim3(KVD / 32, HIDDEN / 32), 256, 0, stream>>>(
      Wv, WvT, HIDDEN, KVD, 1.0f);
  transpose_cvt_kernel<<<dim3(HIDDEN / 32, HIDDEN / 32), 256, 0, stream>>>(
      Wo, WoT, HIDDEN, HIDDEN, 1.0f);

  // 2) Q/K/V projections (Q,K bf16 row-major; V bf16 transposed per batch)
  gemm_bf16_tn<1><<<dim3(HIDDEN / 128, (int)MS / 64), 256, 0, stream>>>(
      Xb, WqT, Qb, (int)MS, HIDDEN, HIDDEN);
  gemm_bf16_tn<1><<<dim3(KVD / 128, (int)MS / 64), 256, 0, stream>>>(
      Xb, WkT, Kb, (int)MS, KVD, HIDDEN);
  gemm_bf16_tn<2><<<dim3(KVD / 128, (int)MS / 64), 256, 0, stream>>>(
      Xb, WvT, Vtr, (int)MS, KVD, HIDDEN);

  // 3) RoPE on Q and K
  rope_kernel<<<(BSZ * SEQ * (NH + NKV) * 64) / 256, 256, 0, stream>>>(Qb, Kb);

  // 4) causal flash attention (async double-buffered K/V staging)
  flash_attn_kernel<<<dim3(SEQ / 128, NH, BSZ), 256, 0, stream>>>(Qb, Kb, Vtr,
                                                                  Ob);

  // 5) output projection (f32 out)
  gemm_bf16_tn<0><<<dim3(HIDDEN / 128, (int)MS / 64), 256, 0, stream>>>(
      Ob, WoT, d_out, (int)MS, HIDDEN, HIDDEN);
}
